// DIMKT_90331752169956
// MI455X (gfx1250) — compile-verified, matrix operand order checked by reading, not run on hardware
//
#include <hip/hip_runtime.h>
#include <math.h>

// ---------------------------------------------------------------------------
// DIMKT on gfx1250: precompute (gather + WMMA GEMMs) + persistent WMMA scan.
// ---------------------------------------------------------------------------

typedef __attribute__((ext_vector_type(16))) _Float16 v16h_t;
typedef __attribute__((ext_vector_type(8)))  _Float16 v8h_t;
typedef __attribute__((ext_vector_type(8)))  float    v8f_t;

static constexpr int Bn  = 128;
static constexpr int Sn  = 500;
static constexpr int Dn  = 128;
static constexpr int ROWS = Bn * Sn;      // 64000
static constexpr int WP  = 136;           // LDS f16 row pitch (16B-aligned, bank-spread)

__device__ __forceinline__ float sigf(float x) { return 1.0f / (1.0f + __expf(-x)); }

// Load a 16-f16 WMMA fragment: two contiguous 8-f16 chunks 16 elems apart.
// Lane layout (ISA 7.12.2, 16-bit operands): lanes 0-15 -> K {0..7,16..23},
// lanes 16-31 -> K {8..15,24..31}; caller bakes the per-lane row/kh offsets in.
__device__ __forceinline__ v16h_t ldfrag(const _Float16* p) {
  v8h_t lo = *(const v8h_t*)p;
  v8h_t hi = *(const v8h_t*)(p + 16);
  return __builtin_shufflevector(lo, hi, 0,1,2,3,4,5,6,7,8,9,10,11,12,13,14,15);
}

__device__ __forceinline__ v8f_t wmma16(v16h_t a, v16h_t b, v8f_t c) {
  return __builtin_amdgcn_wmma_f32_16x16x32_f16(false, a, false, b, (short)0, c,
                                                false, false);
}

// ---------------------------------------------------------------------------
// f32 -> f16 weight conversion
// ---------------------------------------------------------------------------
__global__ void wconv_kernel(const float* __restrict__ src,
                             _Float16* __restrict__ dst, int n) {
  int i = blockIdx.x * 256 + threadIdx.x;
  if (i < n) dst[i] = (_Float16)src[i];
}

// ---------------------------------------------------------------------------
// Embedding gather: builds xcat = [qe,ce,qde,cde] (f16, K=512) and
// fcat = [coe,qde,cde] (f16, K=384). One block per (b,s), 128 threads.
// ---------------------------------------------------------------------------
__global__ void gather_kernel(const int* __restrict__ qseq,
                              const int* __restrict__ qdseq,
                              const int* __restrict__ cseq,
                              const int* __restrict__ q2c,
                              const int* __restrict__ q2cm,
                              const int* __restrict__ q2cd,
                              const float* __restrict__ Eq,
                              const float* __restrict__ Ec,
                              const float* __restrict__ Eqd,
                              const float* __restrict__ Ecd,
                              const float* __restrict__ Eco,
                              _Float16* __restrict__ xcat,
                              _Float16* __restrict__ fcat) {
  int bs = blockIdx.x;
  int d  = threadIdx.x;
  int q  = qseq[bs];
  int qd = qdseq[bs];
  int co = cseq[bs];

  float ce = 0.f, cde = 0.f, msum = 0.f;
#pragma unroll
  for (int j = 0; j < 4; ++j) {
    float m = (float)q2cm[q * 4 + j];
    msum += m;
    ce  += m * Ec [q2c [q * 4 + j] * Dn + d];
    cde += m * Ecd[q2cd[q * 4 + j] * Dn + d];
  }
  float denom = fmaxf(msum, 1.0f);
  ce /= denom; cde /= denom;

  float qe  = Eq [q  * Dn + d];
  float qde = Eqd[qd * Dn + d];
  float coe = Eco[co * Dn + d];

  long xb = (long)bs * 512;
  xcat[xb + 0   + d] = (_Float16)qe;
  xcat[xb + 128 + d] = (_Float16)ce;
  xcat[xb + 256 + d] = (_Float16)qde;
  xcat[xb + 384 + d] = (_Float16)cde;
  long fb = (long)bs * 384;
  fcat[fb + 0   + d] = (_Float16)coe;
  fcat[fb + 128 + d] = (_Float16)qde;
  fcat[fb + 256 + d] = (_Float16)cde;
}

// ---------------------------------------------------------------------------
// Generic WMMA GEMM: C[M,128] = A[M,K](f16) * W[128,ldb](f16, row n = out col)
// + bias. Block = 8 waves, each wave one 16x16 C tile; grid.x = M/16.
// ---------------------------------------------------------------------------
__global__ void gemm16_kernel(const _Float16* __restrict__ A, int lda, int K,
                              const _Float16* __restrict__ Wt, int ldb,
                              const float* __restrict__ bias,
                              float* __restrict__ C, int ldc) {
  int w    = threadIdx.x >> 5;
  int lane = threadIdx.x & 31;
  int m0   = blockIdx.x * 16;
  int n0   = w * 16;
  int r    = lane & 15;
  int kh   = (lane >> 4) * 8;

  const _Float16* arow = A  + (long)(m0 + r) * lda + kh;
  const _Float16* brow = Wt + (long)(n0 + r) * ldb + kh;

  v8f_t acc = {};
  for (int k0 = 0; k0 < K; k0 += 32) {
    v16h_t a = ldfrag(arow + k0);
    v16h_t b = ldfrag(brow + k0);
    acc = wmma16(a, b, acc);
  }

  int nl = lane & 15;
  int mh = (lane >> 4) * 8;
  float bv = bias[n0 + nl];
#pragma unroll
  for (int i = 0; i < 8; ++i)
    C[(long)(m0 + mh + i) * ldc + n0 + nl] = acc[i] + bv;
}

// ---------------------------------------------------------------------------
// Persistent recurrence. 8 blocks x 256 threads; block owns 16 batch rows.
// All 5 recurrent weight matrices resident in LDS (f16, pitch WP).
// ---------------------------------------------------------------------------
static constexpr size_t SCAN_LDS_BYTES =
    (size_t)(5 * 128 * WP + 3 * 16 * WP) * 2 + (16 * 128 + 16) * 4;  // 195392

__global__ void scan_kernel(const float* __restrict__ xall,
                            const float* __restrict__ kfix,
                            const float* __restrict__ pfix1,
                            const float* __restrict__ pfix2,
                            const _Float16* __restrict__ Ws1H,
                            const _Float16* __restrict__ Ws2H,
                            const _Float16* __restrict__ Wp1H,
                            const _Float16* __restrict__ Wp2H,
                            const _Float16* __restrict__ WkiH,
                            const float* __restrict__ bsd1,
                            const float* __restrict__ bsd2,
                            const float* __restrict__ bpk1,
                            const float* __restrict__ bpk2,
                            const float* __restrict__ h0,
                            float* __restrict__ y) {
  extern __shared__ char smem[];
  _Float16* ws1 = (_Float16*)smem;          // 128*WP each
  _Float16* ws2 = ws1 + 128 * WP;
  _Float16* wp1 = ws2 + 128 * WP;
  _Float16* wp2 = wp1 + 128 * WP;
  _Float16* wkh = wp2 + 128 * WP;
  _Float16* aI  = wkh + 128 * WP;           // 16*WP staging each
  _Float16* aS  = aI  + 16 * WP;
  _Float16* aH  = aS  + 16 * WP;
  float*    hS   = (float*)(aH + 16 * WP);  // 16*128 f32
  float*    yacc = hS + 16 * 128;           // 16 f32

  int tid = threadIdx.x;
  int b0  = blockIdx.x * 16;

  // Preload weights (f16) into LDS with padded pitch.
  for (int e = tid; e < 128 * 128; e += 256) {
    int rr = e >> 7, cc = e & 127;
    ws1[rr * WP + cc] = Ws1H[rr * 128 + cc];
    ws2[rr * WP + cc] = Ws2H[rr * 128 + cc];
    wp1[rr * WP + cc] = Wp1H[rr * 256 + cc];   // sdf part of W_pka1
    wp2[rr * WP + cc] = Wp2H[rr * 256 + cc];   // sdf part of W_pka2
    wkh[rr * WP + cc] = WkiH[rr * 512 + cc];   // h part of W_ki
  }
  for (int e = tid; e < 16 * 128; e += 256) {
    float hv = h0[(b0 + (e >> 7)) * Dn + (e & 127)];
    hS[e] = hv;
    aH[(e >> 7) * WP + (e & 127)] = (_Float16)hv;  // f16 copy for ki GEMM
  }
  __syncthreads();

  int w    = tid >> 5;
  int lane = tid & 31;
  int n0   = w * 16;
  int nl   = lane & 15;
  int nG   = n0 + nl;
  int mh   = (lane >> 4) * 8;   // C-fragment M offset AND fragment K-half offset
  float vb1 = bsd1[nG], vb2 = bsd2[nG], vp1 = bpk1[nG], vp2 = bpk2[nG];

  const _Float16* pAI = aI  + nl * WP + mh;
  const _Float16* pAH = aH  + nl * WP + mh;
  const _Float16* pAS = aS  + nl * WP + mh;
  const _Float16* pB1 = ws1 + (n0 + nl) * WP + mh;
  const _Float16* pB2 = ws2 + (n0 + nl) * WP + mh;
  const _Float16* pBP1 = wp1 + (n0 + nl) * WP + mh;
  const _Float16* pBP2 = wp2 + (n0 + nl) * WP + mh;
  const _Float16* pBK  = wkh + (n0 + nl) * WP + mh;

  for (int t = 0; t < Sn - 1; ++t) {
    // phase 0: stage isdf = x_t - h (f16); zero y accumulator
    for (int e = tid; e < 16 * 128; e += 256) {
      int m = e >> 7, n = e & 127;
      float xv = xall[(((long)(b0 + m)) * Sn + t) * Dn + n];
      aI[m * WP + n] = (_Float16)(xv - hS[m * 128 + n]);
    }
    if (tid < 16) yacc[tid] = 0.0f;
    __syncthreads();

    // phase 1: sdf1/sdf2 GEMMs on isdf, ki GEMM on h
    v8f_t a1 = {}, a2 = {}, a5 = {};
#pragma unroll
    for (int k0 = 0; k0 < 128; k0 += 32) {
      v16h_t fa  = ldfrag(pAI + k0);
      v16h_t fah = ldfrag(pAH + k0);
      a1 = wmma16(fa,  ldfrag(pB1 + k0), a1);
      a2 = wmma16(fa,  ldfrag(pB2 + k0), a2);
      a5 = wmma16(fah, ldfrag(pBK + k0), a5);
    }
#pragma unroll
    for (int i = 0; i < 8; ++i) {
      int m = mh + i;
      float s = sigf(a1[i] + vb1) * tanhf(a2[i] + vb2);
      aS[m * WP + nG] = (_Float16)s;
    }
    __syncthreads();

    // phase 2: pka GEMMs on sdf, then gate/update/dot; also refresh f16 h
    v8f_t a3 = {}, a4 = {};
#pragma unroll
    for (int k0 = 0; k0 < 128; k0 += 32) {
      v16h_t fs = ldfrag(pAS + k0);
      a3 = wmma16(fs, ldfrag(pBP1 + k0), a3);
      a4 = wmma16(fs, ldfrag(pBP2 + k0), a4);
    }
#pragma unroll
    for (int i = 0; i < 8; ++i) {
      int m = mh + i;
      long gi = (((long)(b0 + m)) * Sn + t) * Dn + nG;
      float g  = sigf(a5[i] + kfix[gi]);
      float pk = sigf(a3[i] + pfix1[gi] + vp1) * tanhf(a4[i] + pfix2[gi] + vp2);
      float hv = hS[m * 128 + nG];
      float hn = g * hv + (1.0f - g) * pk;
      hS[m * 128 + nG] = hn;                 // lane exclusively owns (m,nG)
      aH[m * WP + nG]  = (_Float16)hn;       // f16 copy for next step's ki GEMM
      float xn = xall[(((long)(b0 + m)) * Sn + (t + 1)) * Dn + nG];
      float r = xn * hn;
      // reduce across the 16 lanes sharing row m (masks stay within each half)
      r += __shfl_xor(r, 1);
      r += __shfl_xor(r, 2);
      r += __shfl_xor(r, 4);
      r += __shfl_xor(r, 8);
      if (nl == 0) atomicAdd(&yacc[m], r);
    }
    __syncthreads();
    if (tid < 16) y[(long)(b0 + tid) * Sn + t] = sigf(yacc[tid]);
  }
  if (tid < 16) y[(long)(b0 + tid) * Sn + (Sn - 1)] = 0.0f;
}

// ---------------------------------------------------------------------------
extern "C" void kernel_launch(void* const* d_in, const int* in_sizes, int n_in,
                              void* d_out, int out_size, void* d_ws, size_t ws_size,
                              hipStream_t stream) {
  (void)in_sizes; (void)n_in; (void)out_size; (void)ws_size;

  const int*   qseq   = (const int*)d_in[0];
  const int*   qdseq  = (const int*)d_in[1];
  const int*   cseq   = (const int*)d_in[2];
  const int*   q2c    = (const int*)d_in[3];
  const int*   q2cd   = (const int*)d_in[4];
  const float* Eq     = (const float*)d_in[5];
  const float* Ec     = (const float*)d_in[6];
  const float* Eqd    = (const float*)d_in[7];
  const float* Ecd    = (const float*)d_in[8];
  const float* Eco    = (const float*)d_in[9];
  const float* Wx     = (const float*)d_in[10];
  const float* bx     = (const float*)d_in[11];
  const float* Wsd1   = (const float*)d_in[12];
  const float* bsd1   = (const float*)d_in[13];
  const float* Wsd2   = (const float*)d_in[14];
  const float* bsd2   = (const float*)d_in[15];
  const float* Wpk1   = (const float*)d_in[16];
  const float* bpk1   = (const float*)d_in[17];
  const float* Wpk2   = (const float*)d_in[18];
  const float* bpk2   = (const float*)d_in[19];
  const float* Wki    = (const float*)d_in[20];
  const float* bki    = (const float*)d_in[21];
  const float* h0     = (const float*)d_in[22];
  const int*   q2cm   = (const int*)d_in[23];   // inserted last in setup_inputs

  float* y = (float*)d_out;

  // workspace layout
  char* ws = (char*)d_ws;
  _Float16* WxH  = (_Float16*)ws;            // 128*512
  _Float16* WkiH = WxH  + 128 * 512;         // 128*512
  _Float16* Wp1H = WkiH + 128 * 512;         // 128*256
  _Float16* Wp2H = Wp1H + 128 * 256;         // 128*256
  _Float16* Ws1H = Wp2H + 128 * 256;         // 128*128
  _Float16* Ws2H = Ws1H + 128 * 128;         // 128*128
  _Float16* xcat = Ws2H + 128 * 128;         // ROWS*512
  _Float16* fcat = xcat + (long)ROWS * 512;  // ROWS*384
  float* xall  = (float*)(fcat + (long)ROWS * 384);  // ROWS*128 each below
  float* kfix  = xall  + (long)ROWS * 128;
  float* pfix1 = kfix  + (long)ROWS * 128;
  float* pfix2 = pfix1 + (long)ROWS * 128;

  // 1) weight conversions to f16
  wconv_kernel<<<(128 * 512 + 255) / 256, 256, 0, stream>>>(Wx,   WxH,  128 * 512);
  wconv_kernel<<<(128 * 512 + 255) / 256, 256, 0, stream>>>(Wki,  WkiH, 128 * 512);
  wconv_kernel<<<(128 * 256 + 255) / 256, 256, 0, stream>>>(Wpk1, Wp1H, 128 * 256);
  wconv_kernel<<<(128 * 256 + 255) / 256, 256, 0, stream>>>(Wpk2, Wp2H, 128 * 256);
  wconv_kernel<<<(128 * 128 + 255) / 256, 256, 0, stream>>>(Wsd1, Ws1H, 128 * 128);
  wconv_kernel<<<(128 * 128 + 255) / 256, 256, 0, stream>>>(Wsd2, Ws2H, 128 * 128);

  // 2) embedding gather into xcat / fcat
  gather_kernel<<<ROWS, 128, 0, stream>>>(qseq, qdseq, cseq, q2c, q2cm, q2cd,
                                          Eq, Ec, Eqd, Ecd, Eco, xcat, fcat);

  // 3) bulk WMMA GEMMs (x_all + time-invariant gate contributions)
  gemm16_kernel<<<ROWS / 16, 256, 0, stream>>>(xcat, 512, 512, WxH,        512, bx,   xall,  128);
  gemm16_kernel<<<ROWS / 16, 256, 0, stream>>>(fcat, 384, 384, WkiH + 128, 512, bki,  kfix,  128);
  gemm16_kernel<<<ROWS / 16, 256, 0, stream>>>(fcat, 384, 128, Wp1H + 128, 256, bpk1, pfix1, 128);
  gemm16_kernel<<<ROWS / 16, 256, 0, stream>>>(fcat, 384, 128, Wp2H + 128, 256, bpk2, pfix2, 128);

  // 4) persistent recurrence (needs >64KB dynamic LDS). Idempotent host-side
  //    attribute set each call (no static guards; capture-safe, not a stream op).
  hipFuncSetAttribute(reinterpret_cast<const void*>(scan_kernel),
                      hipFuncAttributeMaxDynamicSharedMemorySize,
                      (int)SCAN_LDS_BYTES);
  scan_kernel<<<Bn / 16, 256, SCAN_LDS_BYTES, stream>>>(
      xall, kfix, pfix1, pfix2, Ws1H, Ws2H, Wp1H, Wp2H, WkiH,
      bsd1, bsd2, bpk1, bpk2, h0, y);
}
